// GAE_52742198395357
// MI455X (gfx1250) — matrix operand order repaired and checked by reading, not compiled, not run on hardware
//
#include <hip/hip_runtime.h>
#include <hip/hip_bf16.h>

// Problem sizes (fixed by the reference)
#define N_NODES 16384
#define E_EDGES 262144
#define F_IN    1024
#define H1_DIM  256
#define NZ_DIM  64

// ---------------------------------------------------------------------------
// WMMA fragment types (CDNA5 wave32, v_wmma_f32_16x16x32_bf16)
// ---------------------------------------------------------------------------
typedef __attribute__((ext_vector_type(16))) __bf16          v16bf;
typedef __attribute__((ext_vector_type(8)))  float           v8f;
typedef __attribute__((ext_vector_type(16))) unsigned short  v16us;

union BF16Frag {
    v16bf bf;
    v16us us;
    uint4 q[2];
};

// float -> bf16 bits, round-to-nearest-even
__device__ __forceinline__ unsigned short f2bf_bits(float f) {
    unsigned u = __float_as_uint(f);
    u += 0x7FFFu + ((u >> 16) & 1u);
    return (unsigned short)(u >> 16);
}

// ---------------------------------------------------------------------------
// Pack kernels: convert f32 operands to bf16 once, outside the WMMA K-loops.
// ---------------------------------------------------------------------------
__global__ void pack_bf16(const float* __restrict__ src,
                          unsigned short* __restrict__ dst, size_t n)
{
    size_t t = (size_t)blockIdx.x * blockDim.x + threadIdx.x;
    if (t < n) dst[t] = f2bf_bits(src[t]);
}

// src is [K, Nn] row-major f32; dst is [Nn, K] row-major bf16 (i.e. B^T),
// so WMMA B-fragments become contiguous along K.
__global__ void pack_bf16_transpose(const float* __restrict__ src,
                                    unsigned short* __restrict__ dst,
                                    int K, int Nn)
{
    int t = blockIdx.x * blockDim.x + threadIdx.x;
    if (t >= K * Nn) return;
    int n = t / K;
    int k = t - n * K;
    dst[t] = f2bf_bits(src[(size_t)k * Nn + n]);
}

// ---------------------------------------------------------------------------
// Dense GEMM: C[M,Nn] = A[M,K] @ B[K,Nn], with A bf16 row-major and
// B supplied pre-transposed bf16 (Bt[Nn,K] row-major). f32 accumulate.
// One 16x16 output tile per wave; every fragment = two b128 loads per lane.
//
// Fragment maps (ISA 7.12.2, wave32), lane = {half = lane>>4, sub = lane&15}:
//   A 16x32 bf16 : row = sub;  elems 0..7 -> K = half*8 + i,
//                              elems 8..15 -> K = 16 + half*8 + (i-8)
//   B 32x16 bf16 : col = sub;  same K map as A
//   C 16x16 f32  : col = sub;  vgpr r -> row = half*8 + r
// ---------------------------------------------------------------------------
__global__ void wmma_gemm_bf16(const unsigned short* __restrict__ A,
                               const unsigned short* __restrict__ Bt,
                               float* __restrict__ C,
                               int M, int Nn, int K)
{
    const int lane = threadIdx.x & 31;
    const int wave = (int)((blockIdx.x * blockDim.x + threadIdx.x) >> 5);
    const int tilesN = Nn >> 4;
    const int tm = wave / tilesN;
    const int tn = wave - tm * tilesN;
    if (tm * 16 >= M) return;

    const int half = lane >> 4;
    const int sub  = lane & 15;
    const int ncol = tn * 16 + sub;

    const unsigned short* arow = A  + (size_t)(tm * 16 + sub) * K;
    const unsigned short* brow = Bt + (size_t)ncol * K;

    v8f acc = {};
    for (int k0 = 0; k0 < K; k0 += 32) {
        __builtin_prefetch(arow + k0 + 128, 0, 1);   // global_prefetch_b8
        BF16Frag a, b;
        a.q[0] = *(const uint4*)(arow + k0 + half * 8);
        a.q[1] = *(const uint4*)(arow + k0 + 16 + half * 8);
        b.q[0] = *(const uint4*)(brow + k0 + half * 8);
        b.q[1] = *(const uint4*)(brow + k0 + 16 + half * 8);
        acc = __builtin_amdgcn_wmma_f32_16x16x32_bf16(
            false, a.bf, false, b.bf, (short)0, acc, false, false);
    }

#pragma unroll
    for (int r = 0; r < 8; ++r) {
        C[(size_t)(tm * 16 + half * 8 + r) * Nn + ncol] = acc[r];
    }
}

// ---------------------------------------------------------------------------
// a_bar = z @ z^T from pre-packed bf16 z (N_NODES x NZ_DIM). K = 64 = 2 WMMAs.
// B[k][n] = z[n][k], read directly from z rows (z is its own transpose source).
//
// Store path is the bandwidth bottleneck (1.07 GB output): each block (8
// waves) computes a 16x128 strip, stages it in LDS, then emits fully
// coalesced 128-bit global stores (512 contiguous bytes per row per wave).
// ---------------------------------------------------------------------------
__global__ void wmma_zzt(const unsigned short* __restrict__ zb,
                         float* __restrict__ abar)
{
    __shared__ float tileC[16 * 128];                 // 8 KB

    const int lane = threadIdx.x & 31;
    const int wv   = threadIdx.x >> 5;                // 0..7: column tile in block
    const int tm   = blockIdx.x >> 7;                 // row tile      (0..1023)
    const int tng  = blockIdx.x & 127;                // col tile group (0..127)
    const int tn   = tng * 8 + wv;

    const int half = lane >> 4;
    const int sub  = lane & 15;
    const size_t arow = (size_t)(tm * 16 + sub) * NZ_DIM;
    const size_t brow = (size_t)(tn * 16 + sub) * NZ_DIM;

    v8f acc = {};
#pragma unroll
    for (int k0 = 0; k0 < NZ_DIM; k0 += 32) {
        BF16Frag a, b;
        a.q[0] = *(const uint4*)(zb + arow + k0 + half * 8);
        a.q[1] = *(const uint4*)(zb + arow + k0 + 16 + half * 8);
        b.q[0] = *(const uint4*)(zb + brow + k0 + half * 8);
        b.q[1] = *(const uint4*)(zb + brow + k0 + 16 + half * 8);
        acc = __builtin_amdgcn_wmma_f32_16x16x32_bf16(
            false, a.bf, false, b.bf, (short)0, acc, false, false);
    }

    // tile -> LDS (ds_store_b32)
#pragma unroll
    for (int r = 0; r < 8; ++r) {
        tileC[(half * 8 + r) * 128 + wv * 16 + sub] = acc[r];
    }
    __syncthreads();

    // LDS -> global, coalesced b128 stores
    const size_t colbase = (size_t)tng * 128;
#pragma unroll
    for (int j = 0; j < 2; ++j) {
        int idx = (int)threadIdx.x + j * 256;         // 0..511 float4 slots
        int row = idx >> 5;                           // 0..15
        int c4  = idx & 31;                           // 0..31
        float4 v = *(const float4*)(tileC + row * 128 + c4 * 4);
        *(float4*)(abar + (size_t)(tm * 16 + row) * N_NODES + colbase + c4 * 4) = v;
    }
}

// ---------------------------------------------------------------------------
// SpMM scatter: acc[dst,:] += w * sup[src,:]. One thread = one edge x 4 feats.
// ---------------------------------------------------------------------------
__global__ void spmm_scatter(const int* __restrict__ ei,
                             const float* __restrict__ ew,
                             const float* __restrict__ sup,
                             float* __restrict__ acc,
                             int nedges, int d)
{
    const int groups = d >> 2;
    long long t = (long long)blockIdx.x * blockDim.x + threadIdx.x;
    long long total = (long long)nedges * groups;
    if (t >= total) return;
    int e  = (int)(t / groups);
    int f4 = (int)(t - (long long)e * groups) * 4;
    int src = ei[e];
    int dst = ei[nedges + e];
    float w = ew[e];
    const float4 v = *(const float4*)(sup + (size_t)src * d + f4);
    float* o = acc + (size_t)dst * d + f4;
    atomicAdd(o + 0, w * v.x);
    atomicAdd(o + 1, w * v.y);
    atomicAdd(o + 2, w * v.z);
    atomicAdd(o + 3, w * v.w);
}

// ---------------------------------------------------------------------------
// Utility kernels
// ---------------------------------------------------------------------------
__global__ void zero_f32(float* __restrict__ p, size_t n) {
    size_t t = (size_t)blockIdx.x * blockDim.x + threadIdx.x;
    if (t < n) p[t] = 0.0f;
}

// bf16(relu(in)) -- hidden1 pack for GEMM2
__global__ void relu_pack(const float* __restrict__ in,
                          unsigned short* __restrict__ out, size_t n)
{
    size_t t = (size_t)blockIdx.x * blockDim.x + threadIdx.x;
    if (t < n) {
        float v = in[t];
        v = v > 0.0f ? v : 0.0f;
        out[t] = f2bf_bits(v);
    }
}

// z = relu(acc2) -> f32 output tail AND bf16 pack for the z@z^T kernel
__global__ void relu_pack_z(const float* __restrict__ acc2,
                            float* __restrict__ z_out,
                            unsigned short* __restrict__ zb)
{
    size_t t = (size_t)blockIdx.x * blockDim.x + threadIdx.x;
    if (t >= (size_t)N_NODES * NZ_DIM) return;
    float v = acc2[t];
    v = v > 0.0f ? v : 0.0f;
    z_out[t] = v;
    zb[t]    = f2bf_bits(v);
}

// ---------------------------------------------------------------------------
// Host-side launch
// ---------------------------------------------------------------------------
extern "C" void kernel_launch(void* const* d_in, const int* in_sizes, int n_in,
                              void* d_out, int out_size, void* d_ws, size_t ws_size,
                              hipStream_t stream)
{
    const float* x  = (const float*)d_in[0];   // [N, F_IN]
    const float* W1 = (const float*)d_in[1];   // [F_IN, H1]
    const float* W2 = (const float*)d_in[2];   // [H1, NZ]
    const float* ew = (const float*)d_in[3];   // [E]
    const int*   ei = (const int*)d_in[4];     // [2, E] flat: src = ei[e], dst = ei[E+e]

    float* abar  = (float*)d_out;                               // [N, N]
    float* z_out = abar + (size_t)N_NODES * N_NODES;            // [N, NZ]

    // Workspace layout (f32 regions first, then 16B-aligned bf16 regions)
    float* support1 = (float*)d_ws;                              // N*H1   f32
    float* acc1     = support1 + (size_t)N_NODES * H1_DIM;       // N*H1   f32
    float* support2 = acc1     + (size_t)N_NODES * H1_DIM;       // N*NZ   f32
    float* acc2     = support2 + (size_t)N_NODES * NZ_DIM;       // N*NZ   f32
    unsigned short* xb  = (unsigned short*)(acc2 + (size_t)N_NODES * NZ_DIM); // N*F_IN
    unsigned short* w1t = xb  + (size_t)N_NODES * F_IN;          // H1*F_IN (W1^T)
    unsigned short* w2t = w1t + (size_t)H1_DIM * F_IN;           // NZ*H1   (W2^T)
    unsigned short* h1b = w2t + (size_t)NZ_DIM * H1_DIM;         // N*H1
    unsigned short* zb  = h1b + (size_t)N_NODES * H1_DIM;        // N*NZ

    // 0) pack inputs to bf16 (weights transposed for contiguous-K B frags)
    {
        size_t n = (size_t)N_NODES * F_IN;
        pack_bf16<<<(unsigned)((n + 255) / 256), 256, 0, stream>>>(x, xb, n);
        int n1 = F_IN * H1_DIM;
        pack_bf16_transpose<<<(n1 + 255) / 256, 256, 0, stream>>>(W1, w1t, F_IN, H1_DIM);
        int n2 = H1_DIM * NZ_DIM;
        pack_bf16_transpose<<<(n2 + 255) / 256, 256, 0, stream>>>(W2, w2t, H1_DIM, NZ_DIM);
    }

    // 1) zero both accumulators (acc1, then support2/acc2 are past it; only
    //    acc1 and acc2 need zeroing — they are not contiguous now, two calls)
    {
        size_t n1 = (size_t)N_NODES * H1_DIM;
        zero_f32<<<(unsigned)((n1 + 255) / 256), 256, 0, stream>>>(acc1, n1);
        size_t n2 = (size_t)N_NODES * NZ_DIM;
        zero_f32<<<(unsigned)((n2 + 255) / 256), 256, 0, stream>>>(acc2, n2);
    }

    // 2) support1 = x @ W1            [16384,1024] x [1024,256]
    {
        int waves = (N_NODES / 16) * (H1_DIM / 16);              // 16384
        wmma_gemm_bf16<<<waves / 8, 256, 0, stream>>>(xb, w1t, support1,
                                                      N_NODES, H1_DIM, F_IN);
    }

    // 3) acc1[dst] += w * support1[src]
    {
        long long total = (long long)E_EDGES * (H1_DIM / 4);
        spmm_scatter<<<(unsigned)((total + 255) / 256), 256, 0, stream>>>(
            ei, ew, support1, acc1, E_EDGES, H1_DIM);
    }

    // 4) h1b = bf16(relu(acc1))
    {
        size_t n = (size_t)N_NODES * H1_DIM;
        relu_pack<<<(unsigned)((n + 255) / 256), 256, 0, stream>>>(acc1, h1b, n);
    }

    // 5) support2 = hidden1 @ W2      [16384,256] x [256,64]
    {
        int waves = (N_NODES / 16) * (NZ_DIM / 16);              // 4096
        wmma_gemm_bf16<<<waves / 8, 256, 0, stream>>>(h1b, w2t, support2,
                                                      N_NODES, NZ_DIM, H1_DIM);
    }

    // 6) acc2[dst] += w * support2[src]
    {
        long long total = (long long)E_EDGES * (NZ_DIM / 4);
        spmm_scatter<<<(unsigned)((total + 255) / 256), 256, 0, stream>>>(
            ei, ew, support2, acc2, E_EDGES, NZ_DIM);
    }

    // 7) z = relu(acc2) -> output tail (f32) + bf16 pack
    {
        size_t n = (size_t)N_NODES * NZ_DIM;
        relu_pack_z<<<(unsigned)((n + 255) / 256), 256, 0, stream>>>(acc2, z_out, zb);
    }

    // 8) a_bar = z @ z^T              [16384,64] x [64,16384]
    {
        long long blocks = ((long long)(N_NODES / 16) * (N_NODES / 16)) / 8; // 131072
        wmma_zzt<<<(unsigned)blocks, 256, 0, stream>>>(zb, abar);
    }
}
